// PCAClassifier_14104672600628
// MI455X (gfx1250) — matrix operand order repaired and checked by reading, not compiled, not run on hardware
//
#include <hip/hip_runtime.h>
#include <math.h>

// Problem constants (match reference)
#define B_N 8192
#define C_N 200
#define D_N 1024
#define K_N 50
#define KP  64    // K padded to 64 (col/row 50 = mean column, 51..63 = zero)
#define BT  128   // batch tile (rows per workgroup)
#define CT  2     // classes per workgroup (output N = CT*64 = 128)
#define DC  64    // D chunk per LDS stage
#define LST 68    // f32 LDS row stride (floats), breaks 64-bank stride
#define PST 33    // packed-pair LDS row stride (dwords), odd -> conflict-free

typedef __attribute__((ext_vector_type(2)))  float  v2f;
typedef __attribute__((ext_vector_type(8)))  float  v8f;
typedef __attribute__((ext_vector_type(16))) __bf16 v16bf;

union BFrag { unsigned u[8]; v16bf v; };

// --- WMMA wrappers ----------------------------------------------------------
__device__ __forceinline__ v8f wmma_f32(v2f a, v2f b, v8f c) {
  return __builtin_amdgcn_wmma_f32_16x16x4_f32(false, a, false, b, (short)0, c,
                                               false, false);
}
__device__ __forceinline__ v8f wmma_bf16(v16bf a, v16bf b, v8f c) {
  return __builtin_amdgcn_wmma_f32_16x16x32_bf16(false, a, false, b, (short)0,
                                                 c, false, false);
}

// --- bf16 split helpers (round-to-nearest-even) -----------------------------
__device__ __forceinline__ unsigned short f2bf_u(float x) {
  union { float f; unsigned u; } c; c.f = x;
  unsigned u = c.u;
  u += 0x7FFFu + ((u >> 16) & 1u);
  return (unsigned short)(u >> 16);
}
__device__ __forceinline__ float bfu2f(unsigned short s) {
  union { unsigned u; float f; } c; c.u = ((unsigned)s) << 16;
  return c.f;
}
__device__ __forceinline__ void split1(float x, unsigned short& h,
                                       unsigned short& l) {
  h = f2bf_u(x);
  l = f2bf_u(x - bfu2f(h));
}

// ---------------------------------------------------------------------------
// Prep 1: split features into hi/lo bf16 planes (row-major, K-pairs packed)
// ---------------------------------------------------------------------------
__global__ void __launch_bounds__(256) split_feat_kernel(
    const float* __restrict__ f, unsigned short* __restrict__ fH,
    unsigned short* __restrict__ fL) {
  size_t i = (size_t)blockIdx.x * 256 + threadIdx.x;  // over B*D/4
  const float4 v = ((const float4*)f)[i];
  unsigned short h0, l0, h1, l1, h2, l2, h3, l3;
  split1(v.x, h0, l0); split1(v.y, h1, l1);
  split1(v.z, h2, l2); split1(v.w, h3, l3);
  uint2 hp, lp;
  hp.x = (unsigned)h0 | ((unsigned)h1 << 16);
  hp.y = (unsigned)h2 | ((unsigned)h3 << 16);
  lp.x = (unsigned)l0 | ((unsigned)l1 << 16);
  lp.y = (unsigned)l2 | ((unsigned)l3 << 16);
  ((uint2*)fH)[i] = hp;
  ((uint2*)fL)[i] = lp;
}

// ---------------------------------------------------------------------------
// Prep 2: transposed augmented basis WT[c][n][d]: n<50 -> V[:,n], n==50 -> mu,
// else 0; split into hi/lo bf16 planes with d-pairs packed per dword.
// ---------------------------------------------------------------------------
__global__ void __launch_bounds__(256) split_basis_kernel(
    const float* __restrict__ means, const float* __restrict__ pca,
    unsigned short* __restrict__ wH, unsigned short* __restrict__ wL) {
  const int c = blockIdx.x, n = blockIdx.y;
#pragma unroll
  for (int t = 0; t < 2; ++t) {
    const int p = threadIdx.x + t * 256;  // 0..511 pairs
    const int d = 2 * p;
    float x0 = 0.f, x1 = 0.f;
    if (n < K_N) {
      x0 = pca[((size_t)c * D_N + d) * K_N + n];
      x1 = pca[((size_t)c * D_N + d + 1) * K_N + n];
    } else if (n == K_N) {
      x0 = means[(size_t)c * D_N + d];
      x1 = means[(size_t)c * D_N + d + 1];
    }
    unsigned short h0, l0, h1, l1;
    split1(x0, h0, l0);
    split1(x1, h1, l1);
    const size_t o = (((size_t)c * 64 + n) * D_N + d) >> 1;  // dword index
    ((unsigned*)wH)[o] = (unsigned)h0 | ((unsigned)h1 << 16);
    ((unsigned*)wL)[o] = (unsigned)l0 | ((unsigned)l1 << 16);
  }
}

// ---------------------------------------------------------------------------
// Prep 3: f2[b] = ||features[b]||^2 (wave32 per row)
// ---------------------------------------------------------------------------
__global__ void __launch_bounds__(256) f2_kernel(const float* __restrict__ f,
                                                 float* __restrict__ f2) {
  const int wave = threadIdx.x >> 5, lane = threadIdx.x & 31;
  const int row = blockIdx.x * 8 + wave;
  const float4* rp = (const float4*)(f + (size_t)row * D_N);
  float s = 0.f;
#pragma unroll
  for (int it = 0; it < (D_N / 4) / 32; ++it) {
    float4 v = rp[it * 32 + lane];
    s += v.x * v.x + v.y * v.y + v.z * v.z + v.w * v.w;
  }
#pragma unroll
  for (int o = 16; o > 0; o >>= 1) s += __shfl_xor(s, o, 32);
  if (lane == 0) f2[row] = s;
}

// ---------------------------------------------------------------------------
// Prep 4: per-class augmented Gram via f32 WMMA (exact).
// Gaug = W^T W with W = [V | mu]:  Gram (i,j<50), mproj (row 50), m2 (50,50).
// ---------------------------------------------------------------------------
__global__ void __launch_bounds__(256) classprep_kernel(
    const float* __restrict__ means, const float* __restrict__ pca,
    float* __restrict__ gram_pad, float* __restrict__ mproj,
    float* __restrict__ m2) {
  __shared__ float Vt[DC][LST];
  const int cls = blockIdx.x, tid = threadIdx.x;
  const int wave = tid >> 5, lane = tid & 31;
  const int m16 = lane & 15, hi = lane >> 4;
  const int rowbase = (wave & 3) * 16, colbase = (wave >> 2) * 32;

  v8f acc0 = {}, acc1 = {};
  for (int d0 = 0; d0 < D_N; d0 += DC) {
#pragma unroll
    for (int t = 0; t < (DC * KP) / 256; ++t) {
      int i = tid + t * 256;
      int d = i >> 6, k = i & 63;
      float v = 0.f;
      if (k < K_N)       v = pca[((size_t)cls * D_N + (d0 + d)) * K_N + k];
      else if (k == K_N) v = means[(size_t)cls * D_N + d0 + d];
      Vt[d][k] = v;
    }
    __syncthreads();
#pragma unroll
    for (int kk = 0; kk < DC; kk += 4) {
      const int kr = kk + 2 * hi;
      v2f a  = {Vt[kr][rowbase + m16],      Vt[kr + 1][rowbase + m16]};
      v2f b0 = {Vt[kr][colbase + m16],      Vt[kr + 1][colbase + m16]};
      v2f b1 = {Vt[kr][colbase + 16 + m16], Vt[kr + 1][colbase + 16 + m16]};
      acc0 = wmma_f32(a, b0, acc0);
      acc1 = wmma_f32(a, b1, acc1);
    }
    __syncthreads();
  }
#pragma unroll
  for (int t = 0; t < 2; ++t) {
    const int col = colbase + 16 * t + m16;
    v8f acc = t ? acc1 : acc0;
#pragma unroll
    for (int r = 0; r < 8; ++r) {
      const int row = rowbase + r + 8 * hi;
      const float v = acc[r];
      gram_pad[((size_t)cls * 64 + row) * 64 + col] =
          (row < K_N && col < K_N) ? v : 0.f;
      if (row == K_N && col < K_N) mproj[cls * 64 + col] = v;
      if (row == K_N && col == K_N) m2[cls] = v;
    }
  }
}

// ---------------------------------------------------------------------------
// Main kernel. Grid (B/128, C/2). 256 threads = 8 waves.
// GEMM1: 128x128 output tile (128 batch rows x 2 classes of 64 padded cols)
// on split-bf16 WMMA; wave (wM=wave&3, ct=wave>>2) owns a 32x64 slice of its
// class -> 8 accumulators. Intensity: 24 WMMA per 96 LDS dwords per sub-chunk.
// Phase 2 per class (all 8 waves): P = FP - mproj, GP = P @ Gram (f32 WMMA),
// then distance epilogue.
// ---------------------------------------------------------------------------
__global__ void __launch_bounds__(256) pca_main_kernel(
    const unsigned* __restrict__ fH, const unsigned* __restrict__ fL,
    const unsigned* __restrict__ wH, const unsigned* __restrict__ wL,
    const float* __restrict__ gram_pad, const float* __restrict__ mproj,
    const float* __restrict__ m2, const float* __restrict__ f2,
    float* __restrict__ out) {
  // Phase-aliased LDS: GEMM1 split tiles (67584 B), then f32 P/G/GP (87040 B)
  __shared__ __align__(16) unsigned char smem[87040];
  unsigned (*FtH)[PST] = (unsigned (*)[PST])(smem);            // [128][33]
  unsigned (*FtL)[PST] = (unsigned (*)[PST])(smem + 16896);    // [128][33]
  unsigned (*WnH)[PST] = (unsigned (*)[PST])(smem + 33792);    // [128][33]
  unsigned (*WnL)[PST] = (unsigned (*)[PST])(smem + 50688);    // [128][33]
  float (*Pt)[LST]  = (float (*)[LST])(smem);                  // [128][68]
  float (*Gt)[LST]  = (float (*)[LST])(smem + 34816);          // [64][68]
  float (*GPt)[LST] = (float (*)[LST])(smem + 52224);          // [128][68]
  __shared__ float mpr[CT][64];
  __shared__ float sM2[CT];

  const int cls0 = blockIdx.y * CT;
  const int b0   = blockIdx.x * BT;
  const int tid  = threadIdx.x;
  const int wave = tid >> 5, lane = tid & 31;
  const int m16 = lane & 15, hi = lane >> 4;
  const int wM = wave & 3;       // 4 row groups of 32
  const int wC = wave >> 2;      // class owned by this wave in GEMM1
  const int rowbase = 32 * wM;

  if (tid < CT * 64) mpr[tid >> 6][tid & 63] = mproj[(cls0 + (tid >> 6)) * 64 + (tid & 63)];
  if (tid < CT) sM2[tid] = m2[cls0 + tid];

  const int DW = D_N / 2;  // packed dwords per row
  v8f acc[2][4];
#pragma unroll
  for (int mi = 0; mi < 2; ++mi)
#pragma unroll
    for (int ni = 0; ni < 4; ++ni) acc[mi][ni] = (v8f){};

  for (int d0 = 0; d0 < D_N; d0 += DC) {
    const int pg = d0 >> 1;
    // Stage tiles: F (128x32 dwords) and W (2 classes x 64 rows x 32 dwords)
#pragma unroll
    for (int t = 0; t < 16; ++t) {
      int i = tid + t * 256;  // 0..4095
      int r = i >> 5, p = i & 31;
      FtH[r][p] = fH[(size_t)(b0 + r) * DW + pg + p];
      FtL[r][p] = fL[(size_t)(b0 + r) * DW + pg + p];
      WnH[r][p] = wH[((size_t)cls0 * 64 + r) * DW + pg + p];
      WnL[r][p] = wL[((size_t)cls0 * 64 + r) * DW + pg + p];
    }
    __syncthreads();
#pragma unroll
    for (int sub = 0; sub < 2; ++sub) {
      const int pb = sub * 16;
      BFrag aH0, aL0, aH1, aL1;
#pragma unroll
      for (int j = 0; j < 8; ++j) {
        // A 16x32 bf16 layout: pair p = 8*(j>=4) + 4*hi + (j&3)
        const int pa = pb + ((j >> 2) << 3) + (hi << 2) + (j & 3);
        aH0.u[j] = FtH[rowbase + m16][pa];
        aL0.u[j] = FtL[rowbase + m16][pa];
        aH1.u[j] = FtH[rowbase + 16 + m16][pa];
        aL1.u[j] = FtL[rowbase + 16 + m16][pa];
      }
#pragma unroll
      for (int ni = 0; ni < 4; ++ni) {
        BFrag bHf, bLf;
        const int wrow = wC * 64 + 16 * ni + m16;  // W row = class*64 + col
#pragma unroll
        for (int j = 0; j < 8; ++j) {
          // B 32x16 bf16 layout: pair p = 8*(j>=4) + 4*hi + (j&3), N = lane%16
          const int pbj = pb + ((j >> 2) << 3) + (hi << 2) + (j & 3);
          bHf.u[j] = WnH[wrow][pbj];
          bLf.u[j] = WnL[wrow][pbj];
        }
        acc[0][ni] = wmma_bf16(aH0.v, bHf.v, acc[0][ni]);
        acc[0][ni] = wmma_bf16(aH0.v, bLf.v, acc[0][ni]);
        acc[0][ni] = wmma_bf16(aL0.v, bHf.v, acc[0][ni]);
        acc[1][ni] = wmma_bf16(aH1.v, bHf.v, acc[1][ni]);
        acc[1][ni] = wmma_bf16(aH1.v, bLf.v, acc[1][ni]);
        acc[1][ni] = wmma_bf16(aL1.v, bHf.v, acc[1][ni]);
      }
    }
    __syncthreads();
  }

  // ---- Phase 2, sequential per class (all 8 waves cooperate) --------------
  for (int ct = 0; ct < CT; ++ct) {
    // Waves owning this class write P = FP - mproj (col 50 keeps fm; 51..63=0)
    if (wC == ct) {
#pragma unroll
      for (int mi = 0; mi < 2; ++mi)
#pragma unroll
        for (int ni = 0; ni < 4; ++ni) {
          const int col = 16 * ni + m16;
          const float sub = (col < K_N) ? mpr[ct][col] : 0.f;
#pragma unroll
          for (int r = 0; r < 8; ++r)
            Pt[rowbase + 16 * mi + r + 8 * hi][col] = acc[mi][ni][r] - sub;
        }
    }
    // Stage zero-padded Gram for this class
#pragma unroll
    for (int t = 0; t < 16; ++t) {
      int i = tid + t * 256;
      int j = i >> 6, k = i & 63;
      Gt[j][k] = gram_pad[((size_t)(cls0 + ct) * 64 + j) * 64 + k];
    }
    __syncthreads();

    // GEMM2: GP = P @ Gram (f32 WMMA). Wave = one 16x64 row stripe.
    const int rb2 = 16 * wave;
    v8f q[4];
#pragma unroll
    for (int ni = 0; ni < 4; ++ni) q[ni] = (v8f){};
#pragma unroll
    for (int kk = 0; kk < KP; kk += 4) {
      const int kr = kk + 2 * hi;
      v2f a = {Pt[rb2 + m16][kr], Pt[rb2 + m16][kr + 1]};
#pragma unroll
      for (int ni = 0; ni < 4; ++ni) {
        v2f b = {Gt[kr][16 * ni + m16], Gt[kr + 1][16 * ni + m16]};
        q[ni] = wmma_f32(a, b, q[ni]);
      }
    }
#pragma unroll
    for (int ni = 0; ni < 4; ++ni) {
      const int col = 16 * ni + m16;
#pragma unroll
      for (int r = 0; r < 8; ++r)
        GPt[rb2 + r + 8 * hi][col] = q[ni][r];
    }
    __syncthreads();

    // Distance epilogue: one thread per batch row of the tile
    if (tid < BT) {
      float quad = 0.f, s2 = 0.f;
#pragma unroll 8
      for (int k = 0; k < KP; ++k) {
        const float p = Pt[tid][k];
        quad += p * GPt[tid][k];   // GP cols >= 50 are exactly 0
        if (k < K_N) s2 += p * p;  // exclude fm column
      }
      const float fm = Pt[tid][K_N];
      const float d2 = f2[b0 + tid] - 2.f * fm + sM2[ct] - 2.f * s2 + quad;
      out[(size_t)(b0 + tid) * C_N + cls0 + ct] = -sqrtf(fmaxf(d2, 0.f));
    }
    __syncthreads();  // before Pt/Gt/GPt reuse by next class
  }
}

// ---------------------------------------------------------------------------
extern "C" void kernel_launch(void* const* d_in, const int* in_sizes, int n_in,
                              void* d_out, int out_size, void* d_ws,
                              size_t ws_size, hipStream_t stream) {
  const float* features = (const float*)d_in[0];  // [B, D]
  const float* means    = (const float*)d_in[1];  // [C, D]
  const float* pca      = (const float*)d_in[2];  // [C, D, K]
  float* out = (float*)d_out;                     // [B, C]

  // Workspace layout (16B-aligned regions):
  unsigned char* ws = (unsigned char*)d_ws;
  float* gram_pad = (float*)ws;                          // C*64*64 f32
  ws += (size_t)C_N * 64 * 64 * 4;
  float* mprojb = (float*)ws;  ws += (size_t)C_N * 64 * 4;   // C*64 f32
  float* m2b    = (float*)ws;  ws += 1024;                   // C f32 (padded)
  float* f2b    = (float*)ws;  ws += (size_t)B_N * 4;        // B f32
  unsigned short* featH = (unsigned short*)ws;  ws += (size_t)B_N * D_N * 2;
  unsigned short* featL = (unsigned short*)ws;  ws += (size_t)B_N * D_N * 2;
  unsigned short* wtH = (unsigned short*)ws;  ws += (size_t)C_N * 64 * D_N * 2;
  unsigned short* wtL = (unsigned short*)ws;

  split_feat_kernel<<<(B_N * D_N / 4) / 256, 256, 0, stream>>>(features, featH,
                                                               featL);
  split_basis_kernel<<<dim3(C_N, 64), 256, 0, stream>>>(means, pca, wtH, wtL);
  f2_kernel<<<B_N / 8, 256, 0, stream>>>(features, f2b);
  classprep_kernel<<<C_N, 256, 0, stream>>>(means, pca, gram_pad, mprojb, m2b);

  dim3 grid(B_N / BT, C_N / CT);
  pca_main_kernel<<<grid, 256, 0, stream>>>(
      (const unsigned*)featH, (const unsigned*)featL, (const unsigned*)wtH,
      (const unsigned*)wtL, gram_pad, mprojb, m2b, f2b, out);
}
